// QuantizedEmbedding_49460843381149
// MI455X (gfx1250) — matrix operand order repaired and checked by reading, not compiled, not run on hardware
//
#include <hip/hip_runtime.h>

// ---------------- problem constants ----------------
#define K_CODES   8192
#define D_DIM     256
#define N_ROWS    32768            // B*T = 32*1024
#define GAMMA     0.99f
#define ONE_M_G   0.01f            // float32(1.0 - 0.99)
#define EPSV      1e-5f

// WMMA tile config
#define MT   128                   // rows per workgroup (8 waves * 16)
#define KT   64                    // codes per K-tile
#define NKT  (K_CODES / KT)        // 128 iterations
#define ROWDW 132                  // 128 dwords/row + 4 dwords TDM pad (bank stride 4)
#define BUFDW (KT * ROWDW)         // 8448 dwords = 33792 B per buffer

typedef __attribute__((ext_vector_type(16))) _Float16 v16h;
typedef __attribute__((ext_vector_type(8)))  float    v8f;
typedef unsigned int v4u __attribute__((ext_vector_type(4)));
typedef int          v8i __attribute__((ext_vector_type(8)));
typedef int          v4i __attribute__((ext_vector_type(4)));

union HFrag { uint4 q[2]; v16h h; };

// ---------------------------------------------------------------
// Prep: embedW (fp32) -> f16 copy + per-code squared norms
// ---------------------------------------------------------------
__global__ void __launch_bounds__(256) vq_prep_kernel(
    const float* __restrict__ W, _Float16* __restrict__ Wh, float* __restrict__ wsq)
{
  const int wave = threadIdx.x >> 5;
  const int lane = threadIdx.x & 31;
  const int k = blockIdx.x * 8 + wave;
  const float* row = W + (size_t)k * D_DIM;
  const int d0 = lane * 8;
  float4 a = *(const float4*)(row + d0);
  float4 b = *(const float4*)(row + d0 + 4);
  float ss = a.x*a.x + a.y*a.y + a.z*a.z + a.w*a.w
           + b.x*b.x + b.y*b.y + b.z*b.z + b.w*b.w;
  union { _Float16 h[8]; uint4 u; } pk;
  pk.h[0] = (_Float16)a.x; pk.h[1] = (_Float16)a.y;
  pk.h[2] = (_Float16)a.z; pk.h[3] = (_Float16)a.w;
  pk.h[4] = (_Float16)b.x; pk.h[5] = (_Float16)b.y;
  pk.h[6] = (_Float16)b.z; pk.h[7] = (_Float16)b.w;
  *(uint4*)(Wh + (size_t)k * D_DIM + d0) = pk.u;
  #pragma unroll
  for (int m = 16; m >= 1; m >>= 1) ss += __shfl_xor(ss, m, 32);
  if (lane == 0) wsq[k] = ss;
}

// ---------------------------------------------------------------
// Argmin over codebook via WMMA. dist ~ ||w||^2 - 2 z.w  (||z||^2 dropped)
// A-fragments live in VGPRs for the whole kernel. W tiles are DMA'd by the
// Tensor Data Mover into double-buffered LDS (padded rows -> conflict-free),
// B-fragments are double-buffered in registers across d-steps.
// ---------------------------------------------------------------
__global__ void __launch_bounds__(256) vq_argmin_kernel(
    const float* __restrict__ ze, const _Float16* __restrict__ Wh,
    const float* __restrict__ wsq, int* __restrict__ idxI, float* __restrict__ idxF)
{
  __shared__ unsigned int smem[2][BUFDW];      // 2 * 33792 B = 67584 B

  const int tid  = threadIdx.x;
  const int wave = tid >> 5;
  const int lane = tid & 31;
  const int g    = lane >> 4;       // half-wave group
  const int ln   = lane & 15;       // column / row-in-group id
  const int kb   = g * 8;           // A-fragment K base (ISA 16-bit A layout)

  // ---- A fragments: this lane's ze row, converted to f16, kept in VGPRs ----
  const int mrow = blockIdx.x * MT + wave * 16 + ln;
  const float* zp = ze + (size_t)mrow * D_DIM;
  v16h afrag[8];
  #pragma unroll
  for (int s = 0; s < 8; ++s) {
    const int base = 32 * s + kb;
    float4 f0 = *(const float4*)(zp + base);
    float4 f1 = *(const float4*)(zp + base + 4);
    float4 f2 = *(const float4*)(zp + base + 16);
    float4 f3 = *(const float4*)(zp + base + 20);
    v16h a;
    a[0]=(_Float16)f0.x; a[1]=(_Float16)f0.y; a[2]=(_Float16)f0.z; a[3]=(_Float16)f0.w;
    a[4]=(_Float16)f1.x; a[5]=(_Float16)f1.y; a[6]=(_Float16)f1.z; a[7]=(_Float16)f1.w;
    a[8]=(_Float16)f2.x; a[9]=(_Float16)f2.y; a[10]=(_Float16)f2.z; a[11]=(_Float16)f2.w;
    a[12]=(_Float16)f3.x; a[13]=(_Float16)f3.y; a[14]=(_Float16)f3.z; a[15]=(_Float16)f3.w;
    afrag[s] = a;
  }

  // ---- Tensor Data Mover: one descriptor per 64x256-f16 tile (32KB) ----
  const unsigned lds0 = (unsigned)(unsigned long long)(&smem[0][0]);
  const unsigned long long gbase = (unsigned long long)Wh;

  auto issue_tdm = [&](int tile, int buf) {
    const unsigned long long ga = gbase + (unsigned long long)tile * (KT * D_DIM * 2);
    const unsigned lds = lds0 + (unsigned)buf * (BUFDW * 4);
    v4u g0;
    g0[0] = 1u;                                              // count=1 (valid), user mode
    g0[1] = lds;                                             // lds_addr [63:32]
    g0[2] = (unsigned)(ga & 0xFFFFFFFFu);                    // global_addr lo
    g0[3] = (unsigned)((ga >> 32) & 0x01FFFFFFu) | (2u << 30); // addr[56:32] | type=2
    v8i g1;
    // data_size=4B(code 2) | pad_enable | pad_interval=128dw(code 6) | pad_amount=4dw(code 3)
    g1[0] = (int)((2u << 16) | (1u << 20) | (6u << 22) | (3u << 25));
    g1[1] = (int)(128u << 16);          // tensor_dim0 = 128 dwords (one code row)
    g1[2] = (int)(8192u << 16);         // tensor_dim1 = 8192 rows
    g1[3] = (int)(128u << 16);          // tile_dim0 = 128 dwords
    g1[4] = (int)(64u);                 // tile_dim1 = 64 rows, tile_dim2 unused
    g1[5] = (int)(128u);                // tensor_dim0_stride = 128 dwords (contiguous)
    g1[6] = 0;
    g1[7] = 0;
    v4i gz = {0, 0, 0, 0};
#if defined(__clang_major__) && (__clang_major__ >= 23)
    v8i g4 = {0, 0, 0, 0, 0, 0, 0, 0};
    __builtin_amdgcn_tensor_load_to_lds(g0, g1, gz, gz, g4, 0);
#else
    __builtin_amdgcn_tensor_load_to_lds(g0, g1, gz, gz, 0);
#endif
  };

  // prologue: tile 0 -> buffer 0
  if (wave == 0) {
    issue_tdm(0, 0);
    __builtin_amdgcn_s_wait_tensorcnt(0);
  }
  __syncthreads();

  float best[8]; int bidx[8];
  #pragma unroll
  for (int rr = 0; rr < 8; ++rr) { best[rr] = 3.4e38f; bidx[rr] = 0; }

  for (int kt = 0; kt < NKT; ++kt) {
    const int cur = kt & 1, nxt = cur ^ 1;

    float wv[4];
    #pragma unroll
    for (int ct = 0; ct < 4; ++ct) wv[ct] = wsq[kt * KT + ct * 16 + ln];

    if (wave == 0 && kt + 1 < NKT) issue_tdm(kt + 1, nxt);   // DMA overlaps compute
    if (kt + 2 < NKT)                                        // L2 prefetch of tile k+2
      __builtin_prefetch((const char*)Wh + (size_t)(kt + 2) * (KT * D_DIM * 2) + tid * 128, 0, 1);

    const unsigned int* bufp = &smem[cur][0];
    HFrag bf[2][4];
    {  // preload B-fragment set for s = 0
      const int c0 = 4 * g;
      #pragma unroll
      for (int ct = 0; ct < 4; ++ct) {
        const int rowb = (ct * 16 + ln) * ROWDW;
        bf[0][ct].q[0] = *(const uint4*)&bufp[rowb + c0];
        bf[0][ct].q[1] = *(const uint4*)&bufp[rowb + c0 + 8];
      }
    }

    v8f acc[4] = {};
    #pragma unroll
    for (int s = 0; s < 8; ++s) {
      const int curS = s & 1, nxtS = curS ^ 1;
      if (s < 7) {   // load next d-step's B fragments while WMMAs run
        const int c0 = 16 * (s + 1) + 4 * g;
        #pragma unroll
        for (int ct = 0; ct < 4; ++ct) {
          const int rowb = (ct * 16 + ln) * ROWDW;
          bf[nxtS][ct].q[0] = *(const uint4*)&bufp[rowb + c0];
          bf[nxtS][ct].q[1] = *(const uint4*)&bufp[rowb + c0 + 8];
        }
      }
      #pragma unroll
      for (int ct = 0; ct < 4; ++ct) {
        acc[ct] = __builtin_amdgcn_wmma_f32_16x16x32_f16(
            false, afrag[s], false, bf[curS][ct].h, (short)0, acc[ct], false, false);
      }
    }

    #pragma unroll
    for (int ct = 0; ct < 4; ++ct) {
      const int ncode = kt * KT + ct * 16 + ln;
      #pragma unroll
      for (int rr = 0; rr < 8; ++rr) {
        const float dst = wv[ct] - 2.0f * acc[ct][rr];
        if (dst < best[rr]) { best[rr] = dst; bidx[rr] = ncode; }
      }
    }

    if (wave == 0 && kt + 1 < NKT) __builtin_amdgcn_s_wait_tensorcnt(0);
    __syncthreads();
  }

  // ---- cross-lane argmin within each 16-lane group (rows m = rr + 8*g) ----
  #pragma unroll
  for (int rr = 0; rr < 8; ++rr) {
    float b = best[rr]; int i = bidx[rr];
    #pragma unroll
    for (int m = 1; m <= 8; m <<= 1) {
      const float ob = __shfl_xor(b, m, 32);
      const int   oi = __shfl_xor(i, m, 32);
      if (ob < b || (ob == b && oi < i)) { b = ob; i = oi; }
    }
    if (ln == 0) {
      const int mg = blockIdx.x * MT + wave * 16 + g * 8 + rr;
      idxI[mg] = i;
      idxF[mg] = (float)i;
    }
  }
}

// ---------------------------------------------------------------
// Zero the scatter accumulator regions (mt_new ++ Nt_new, contiguous)
// ---------------------------------------------------------------
__global__ void vq_zero_kernel(float* __restrict__ p, int n)
{
  const int i = blockIdx.x * 256 + threadIdx.x;
  if (i < n) p[i] = 0.0f;
}

// ---------------------------------------------------------------
// Gather zq, commit-loss partials, scatter segment-sums (atomics)
// ---------------------------------------------------------------
__global__ void __launch_bounds__(256) vq_gather_kernel(
    const float* __restrict__ ze, const float* __restrict__ Wold,
    const int* __restrict__ idxI, float* __restrict__ zq,
    float* __restrict__ mtAcc, float* __restrict__ NtAcc,
    float* __restrict__ partials)
{
  __shared__ float ls[8];
  const int wave = threadIdx.x >> 5;
  const int lane = threadIdx.x & 31;
  const int n = blockIdx.x * 8 + wave;
  const int i = idxI[n];
  const int d0 = lane * 8;

  const float* wr = Wold + (size_t)i * D_DIM + d0;
  const float* zr = ze   + (size_t)n * D_DIM + d0;
  float4 w0 = *(const float4*)(wr);
  float4 w1 = *(const float4*)(wr + 4);
  float4 z0 = *(const float4*)(zr);
  float4 z1 = *(const float4*)(zr + 4);

  *(float4*)(zq + (size_t)n * D_DIM + d0)     = w0;
  *(float4*)(zq + (size_t)n * D_DIM + d0 + 4) = w1;

  float cs = (w0.x-z0.x)*(w0.x-z0.x) + (w0.y-z0.y)*(w0.y-z0.y)
           + (w0.z-z0.z)*(w0.z-z0.z) + (w0.w-z0.w)*(w0.w-z0.w)
           + (w1.x-z1.x)*(w1.x-z1.x) + (w1.y-z1.y)*(w1.y-z1.y)
           + (w1.z-z1.z)*(w1.z-z1.z) + (w1.w-z1.w)*(w1.w-z1.w);

  float* ma = mtAcc + (size_t)i * D_DIM + d0;
  atomicAdd(ma + 0, ONE_M_G * z0.x); atomicAdd(ma + 1, ONE_M_G * z0.y);
  atomicAdd(ma + 2, ONE_M_G * z0.z); atomicAdd(ma + 3, ONE_M_G * z0.w);
  atomicAdd(ma + 4, ONE_M_G * z1.x); atomicAdd(ma + 5, ONE_M_G * z1.y);
  atomicAdd(ma + 6, ONE_M_G * z1.z); atomicAdd(ma + 7, ONE_M_G * z1.w);
  if (lane == 0) atomicAdd(NtAcc + i, ONE_M_G);

  #pragma unroll
  for (int m = 16; m >= 1; m >>= 1) cs += __shfl_xor(cs, m, 32);
  if (lane == 0) ls[wave] = cs;
  __syncthreads();
  if (threadIdx.x == 0) {
    float s = 0.0f;
    #pragma unroll
    for (int w = 0; w < 8; ++w) s += ls[w];
    partials[blockIdx.x] = s;
  }
}

// ---------------------------------------------------------------
// Finalize 1 (single block): commit mean, Nt_new = g*Nt + (1-g)nt, n = sum
// ---------------------------------------------------------------
__global__ void __launch_bounds__(256) vq_fin1_kernel(
    const float* __restrict__ Nt, const float* __restrict__ partials,
    float* __restrict__ NtNew, float* __restrict__ commitOut, float* __restrict__ nOut)
{
  __shared__ float red[256];
  const int t = threadIdx.x;

  float cs = 0.0f;
  for (int p = t; p < 4096; p += 256) cs += partials[p];

  float ns = 0.0f;
  for (int k = t; k < K_CODES; k += 256) {
    const float v = GAMMA * Nt[k] + NtNew[k];   // NtNew holds (1-g)*nt
    NtNew[k] = v;
    ns += v;
  }

  red[t] = cs; __syncthreads();
  for (int s = 128; s >= 1; s >>= 1) { if (t < s) red[t] += red[t + s]; __syncthreads(); }
  if (t == 0) commitOut[0] = red[0] / (float)((size_t)N_ROWS * D_DIM);
  __syncthreads();

  red[t] = ns; __syncthreads();
  for (int s = 128; s >= 1; s >>= 1) { if (t < s) red[t] += red[t + s]; __syncthreads(); }
  if (t == 0) nOut[0] = red[0];
}

// ---------------------------------------------------------------
// Finalize 2: mt_new = g*mt + (1-g)st ; embedW_new = mt_new / Nn
// ---------------------------------------------------------------
__global__ void __launch_bounds__(256) vq_fin2_kernel(
    const float* __restrict__ mt, const float* __restrict__ NtNew,
    const float* __restrict__ nOut, float* __restrict__ mtNew, float* __restrict__ WNew)
{
  const int e = blockIdx.x * 256 + threadIdx.x;   // < K*D
  const int k = e >> 8;
  const float n = nOut[0];
  const float mv = GAMMA * mt[e] + mtNew[e];      // mtNew holds (1-g)*st
  mtNew[e] = mv;
  const float Nn = (NtNew[k] + EPSV) * n / (n + (float)K_CODES * EPSV);
  WNew[e] = mv / Nn;
}

// ---------------------------------------------------------------
extern "C" void kernel_launch(void* const* d_in, const int* in_sizes, int n_in,
                              void* d_out, int out_size, void* d_ws, size_t ws_size,
                              hipStream_t stream) {
  const float* ze     = (const float*)d_in[0];   // [32768, 256]
  const float* embedW = (const float*)d_in[1];   // [8192, 256]
  const float* mt     = (const float*)d_in[2];   // [8192, 256]
  const float* Nt     = (const float*)d_in[3];   // [8192]

  // ---- workspace layout ----
  char* ws = (char*)d_ws;
  _Float16* Wh     = (_Float16*)ws;                                  // 4 MiB
  float*    wsq    = (float*)(ws + 4194304);                         // 32 KiB
  int*      idxI   = (int*)  (ws + 4194304 + 32768);                 // 128 KiB
  float*    parts  = (float*)(ws + 4194304 + 32768 + 131072);        // 16 KiB
  float*    nOut   = parts + 4096;

  // ---- output layout (floats, concatenated in return order) ----
  float* out    = (float*)d_out;
  float* zq     = out;                  // 8388608
  float* commit = out + 8388608;        // 1
  float* idxF   = out + 8388609;        // 32768
  float* WNew   = out + 8421377;        // 2097152
  float* mtNew  = out + 10518529;       // 2097152
  float* NtNew  = out + 12615681;       // 8192  (contiguous after mtNew)

  vq_prep_kernel<<<K_CODES / 8, 256, 0, stream>>>(embedW, Wh, wsq);
  vq_zero_kernel<<<(2105344 + 255) / 256, 256, 0, stream>>>(mtNew, 2105344);
  vq_argmin_kernel<<<N_ROWS / MT, 256, 0, stream>>>(ze, Wh, wsq, idxI, idxF);
  vq_gather_kernel<<<N_ROWS / 8, 256, 0, stream>>>(ze, embedW, idxI, zq, mtNew, NtNew, parts);
  vq_fin1_kernel<<<1, 256, 0, stream>>>(Nt, parts, NtNew, commit, nOut);
  vq_fin2_kernel<<<(K_CODES * D_DIM) / 256, 256, 0, stream>>>(mt, NtNew, nOut, mtNew, WNew);
}
